// apelpEdgeAttributeDecoder_61727269978221
// MI455X (gfx1250) — matrix operand order repaired and checked by reading, not compiled, not run on hardware
//
#include <hip/hip_runtime.h>
#include <hip/hip_bf16.h>

#define N_NODES 100000
#define N_EDGES 1600000
#define HID 128
#define BN_EPS 1e-5f

typedef __attribute__((ext_vector_type(16))) __bf16          v16bf;
typedef __attribute__((ext_vector_type(8)))  float           v8f;
typedef __attribute__((ext_vector_type(8)))  unsigned short  v8us;
typedef __attribute__((ext_vector_type(16))) unsigned short  v16us;

// float -> bf16 bits (round-to-nearest-even)
__device__ __forceinline__ unsigned short bfbits(float f) {
  unsigned u = __builtin_bit_cast(unsigned, f);
  unsigned r = u + 0x7FFFu + ((u >> 16) & 1u);
  return (unsigned short)(r >> 16);
}
__device__ __forceinline__ float bf2f(unsigned short h) {
  return __builtin_bit_cast(float, (unsigned)h << 16);
}
// build a v16bf A-fragment from two 16B-aligned 8-half LDS slots
__device__ __forceinline__ v16bf ldA(const unsigned short* p0, const unsigned short* p1) {
  v8us lo = *(const v8us*)p0;
  v8us hi = *(const v8us*)p1;
  v16us t = __builtin_shufflevector(lo, hi, 0, 1, 2, 3, 4, 5, 6, 7,
                                    8, 9, 10, 11, 12, 13, 14, 15);
  return __builtin_bit_cast(v16bf, t);
}
__device__ __forceinline__ v16bf ldB(const unsigned short* p) {
  return __builtin_bit_cast(v16bf, *(const v16us*)p);
}

// ---------------------------------------------------------------- utilities
__global__ void k_zero(float* __restrict__ p, long n) {
  long i = (long)blockIdx.x * blockDim.x + threadIdx.x;
  long st = (long)gridDim.x * blockDim.x;
  for (; i < n; i += st) p[i] = 0.0f;
}

__global__ void k_f2bf(const float* __restrict__ in, unsigned short* __restrict__ out, long n) {
  long i = (long)blockIdx.x * blockDim.x + threadIdx.x;
  long st = (long)gridDim.x * blockDim.x;
  for (; i < n; i += st) out[i] = bfbits(in[i]);
}

// Pack fp32 weight [K x N] (row-major) into WMMA-B-fragment-native bf16 layout:
// out[((kt*(N/16)+nt)*32 + lane)*16 + i] = bf16(W[kt*32 + (lane>>4)*16 + i, nt*16 + (lane&15)])
__global__ void k_wpack(const float* __restrict__ w, unsigned short* __restrict__ out,
                        int K, int N) {
  long total = (long)K * N;
  long i = (long)blockIdx.x * blockDim.x + threadIdx.x;
  long st = (long)gridDim.x * blockDim.x;
  int ntn = N >> 4;
  for (; i < total; i += st) {
    int ii = (int)i;
    int e = ii & 15;
    int l = (ii >> 4) & 31;
    int tile = ii >> 9;
    int nt = tile % ntn, kt = tile / ntn;
    int k = kt * 32 + (l >> 4) * 16 + e;
    int n = nt * 16 + (l & 15);
    out[i] = bfbits(w[(size_t)k * N + n]);
  }
}

// scatter-add of 3-wide coords along edges: agg3[dst] += coords[src]
__global__ void k_scatter3(const float* __restrict__ coords, const int* __restrict__ ei,
                           float* __restrict__ agg3) {
  long total = (long)N_EDGES * 3;
  long i = (long)blockIdx.x * blockDim.x + threadIdx.x;
  long st = (long)gridDim.x * blockDim.x;
  for (; i < total; i += st) {
    int e = (int)(i / 3), d = (int)(i % 3);
    int s = ei[e], t = ei[N_EDGES + e];
    atomicAdd(&agg3[(size_t)t * 3 + d], coords[(size_t)s * 3 + d]);
  }
}

// first GIN-0 linear (K=3, VALU): out = ((1+eps)*coords + agg3) @ w1 + b1
__global__ void k_lin0(const float* __restrict__ coords, const float* __restrict__ agg3,
                       const float* __restrict__ w1, const float* __restrict__ b1,
                       const float* __restrict__ eps0, float* __restrict__ out) {
  long total = (long)N_NODES * HID;
  long i = (long)blockIdx.x * blockDim.x + threadIdx.x;
  long st = (long)gridDim.x * blockDim.x;
  float ep = 1.0f + eps0[0];
  for (; i < total; i += st) {
    int n = (int)(i >> 7), c = (int)(i & 127);
    float h = b1[c];
    #pragma unroll
    for (int d = 0; d < 3; ++d)
      h += (ep * coords[(size_t)n * 3 + d] + agg3[(size_t)n * 3 + d]) * w1[d * HID + c];
    out[i] = h;
  }
}

// column sums + sums-of-squares (fp32 source)
__global__ void k_colstats(const float* __restrict__ x, int R, float* __restrict__ stats) {
  int c = threadIdx.x & 127;
  int rg = blockIdx.x * (blockDim.x >> 7) + (threadIdx.x >> 7);
  int step = gridDim.x * (blockDim.x >> 7);
  float s = 0.0f, s2 = 0.0f;
  for (int r = rg; r < R; r += step) {
    float v = x[(size_t)r * HID + c];
    s += v; s2 += v * v;
  }
  atomicAdd(&stats[c], s);
  atomicAdd(&stats[HID + c], s2);
}

// column sums + sums-of-squares (bf16 source)
__global__ void k_colstats_bf(const unsigned short* __restrict__ x, int R,
                              float* __restrict__ stats) {
  int c = threadIdx.x & 127;
  int rg = blockIdx.x * (blockDim.x >> 7) + (threadIdx.x >> 7);
  int step = gridDim.x * (blockDim.x >> 7);
  float s = 0.0f, s2 = 0.0f;
  for (int r = rg; r < R; r += step) {
    float v = bf2f(x[(size_t)r * HID + c]);
    s += v; s2 += v * v;
  }
  atomicAdd(&stats[c], s);
  atomicAdd(&stats[HID + c], s2);
}

// in-place BatchNorm (batch stats, biased var) + ReLU (fp32)
__global__ void k_bnrelu(float* __restrict__ x, int R, const float* __restrict__ stats,
                         const float* __restrict__ g, const float* __restrict__ be) {
  long total = (long)R * HID;
  long i = (long)blockIdx.x * blockDim.x + threadIdx.x;
  long st = (long)gridDim.x * blockDim.x;
  float invR = 1.0f / (float)R;
  for (; i < total; i += st) {
    int c = (int)(i & 127);
    float m  = stats[c] * invR;
    float vv = stats[HID + c] * invR - m * m;
    float v  = (x[i] - m) * rsqrtf(vv + BN_EPS) * g[c] + be[c];
    x[i] = fmaxf(v, 0.0f);
  }
}

// scatter-add 128-wide node features along edges (fp32 atomics)
__global__ void k_scatter128(const float* __restrict__ x, const int* __restrict__ ei,
                             float* __restrict__ agg) {
  long total = (long)N_EDGES * HID;
  long i = (long)blockIdx.x * blockDim.x + threadIdx.x;
  long st = (long)gridDim.x * blockDim.x;
  for (; i < total; i += st) {
    int e = (int)(i >> 7), c = (int)(i & 127);
    int s = ei[e], t = ei[N_EDGES + e];
    atomicAdd(&agg[(size_t)t * HID + c], x[(size_t)s * HID + c]);
  }
}

// agg_io = (1+eps)*x + agg_io
__global__ void k_combine(const float* __restrict__ x, float* __restrict__ agg_io,
                          const float* __restrict__ eps) {
  long total = (long)N_NODES * HID;
  long i = (long)blockIdx.x * blockDim.x + threadIdx.x;
  long st = (long)gridDim.x * blockDim.x;
  float ep = 1.0f + eps[0];
  for (; i < total; i += st) agg_io[i] = ep * x[i] + agg_io[i];
}

// ----------------- WMMA GEMM: [Rx128](fp32) @ packed-bf16[128x128] + b -> fp32
__global__ void __launch_bounds__(256) k_gemm128(const float* __restrict__ A,
                                                 const unsigned short* __restrict__ Wpk,
                                                 const float* __restrict__ bias,
                                                 float* __restrict__ out) {
  __shared__ __align__(16) unsigned short As[16][136];  // 272B stride: 16B-aligned, 4-bank skew
  int tid = threadIdx.x;
  int row0 = blockIdx.x * 16;
  for (int i = tid; i < 16 * HID; i += 256) {
    int r = i >> 7, c = i & 127;
    As[r][c] = bfbits(A[(size_t)(row0 + r) * HID + c]);
  }
  __syncthreads();

  int wave = tid >> 5, lane = tid & 31;
  int half = lane >> 4, m = lane & 15, col = lane & 15;
  int colbase = wave << 4;
  v8f acc = {};
  #pragma unroll
  for (int kt = 0; kt < 4; ++kt) {
    int k0 = kt << 5;
    v16bf a = ldA(&As[m][k0 + half * 16], &As[m][k0 + half * 16 + 8]);
    v16bf b = ldB(&Wpk[(size_t)((((kt << 3) + wave) << 5) + lane) << 4]);
    acc = __builtin_amdgcn_wmma_f32_16x16x32_bf16(false, a, false, b,
                                                  (short)0, acc, false, false);
  }
  float bb = bias[colbase + col];
  #pragma unroll
  for (int r = 0; r < 8; ++r) {
    int crow = r + half * 8;
    out[(size_t)(row0 + crow) * HID + colbase + col] = acc[r] + bb;
  }
}

// ---- decoder GEMM-1: gather bf16 [x1[n0]|x1[n1]] @ packed d_w1 + l*w1_row256 + b1 -> bf16
__global__ void __launch_bounds__(256) k_dec_gemm1(const unsigned short* __restrict__ x1bf,
                                                   const float* __restrict__ coords,
                                                   const int* __restrict__ ei,
                                                   const unsigned short* __restrict__ W1pk,
                                                   const float* __restrict__ w1l,  // d_w1 row 256 (fp32)
                                                   const float* __restrict__ b1,
                                                   unsigned short* __restrict__ eh1) {
  __shared__ __align__(16) unsigned short As[16][264];  // 528B stride: 16B-aligned, 4-bank skew
  __shared__ float Ls[16];
  int tid = threadIdx.x;
  int row0 = blockIdx.x * 16;
  for (int i = tid; i < 16 * 256; i += 256) {
    int r = i >> 8, c = i & 255;
    int e = row0 + r;
    int n = (c < HID) ? ei[e] : ei[N_EDGES + e];
    As[r][c] = x1bf[(size_t)n * HID + (c & 127)];
  }
  if (tid < 16) {
    int e = row0 + tid;
    int n0 = ei[e], n1 = ei[N_EDGES + e];
    float dx = coords[(size_t)n1 * 3 + 0] - coords[(size_t)n0 * 3 + 0];
    float dy = coords[(size_t)n1 * 3 + 1] - coords[(size_t)n0 * 3 + 1];
    float dz = coords[(size_t)n1 * 3 + 2] - coords[(size_t)n0 * 3 + 2];
    Ls[tid] = sqrtf(dx * dx + dy * dy + dz * dz);
  }
  __syncthreads();

  int wave = tid >> 5, lane = tid & 31;
  int half = lane >> 4, m = lane & 15, col = lane & 15;
  int colbase = wave << 4;
  v8f acc = {};
  #pragma unroll
  for (int kt = 0; kt < 8; ++kt) {
    int k0 = kt << 5;
    v16bf a = ldA(&As[m][k0 + half * 16], &As[m][k0 + half * 16 + 8]);
    v16bf b = ldB(&W1pk[(size_t)((((kt << 3) + wave) << 5) + lane) << 4]);
    acc = __builtin_amdgcn_wmma_f32_16x16x32_bf16(false, a, false, b,
                                                  (short)0, acc, false, false);
  }
  float wl = w1l[colbase + col];
  float bb = b1[colbase + col];
  #pragma unroll
  for (int r = 0; r < 8; ++r) {
    int crow = r + half * 8;
    eh1[(size_t)(row0 + crow) * HID + colbase + col] = bfbits(acc[r] + Ls[crow] * wl + bb);
  }
}

// --- decoder tail: BN+ReLU -> [16x128]@packed[128x64]+b2,ReLU -> dot w3 -> sigmoid
__global__ void __launch_bounds__(128) k_dec_tail(const unsigned short* __restrict__ eh1,
                                                  const float* __restrict__ stats,
                                                  const float* __restrict__ g,
                                                  const float* __restrict__ be,
                                                  const unsigned short* __restrict__ W2pk,
                                                  const float* __restrict__ b2,
                                                  const float* __restrict__ W3,
                                                  const float* __restrict__ b3,
                                                  float* __restrict__ out) {
  __shared__ __align__(16) unsigned short Hs[16][136];
  __shared__ float H2[16][68];
  int tid = threadIdx.x;
  int row0 = blockIdx.x * 16;
  const float invE = 1.0f / (float)N_EDGES;
  for (int i = tid; i < 16 * HID; i += 128) {
    int r = i >> 7, c = i & 127;
    float mn = stats[c] * invE;
    float vv = stats[HID + c] * invE - mn * mn;
    float v = (bf2f(eh1[(size_t)(row0 + r) * HID + c]) - mn) * rsqrtf(vv + BN_EPS) * g[c] + be[c];
    Hs[r][c] = bfbits(fmaxf(v, 0.0f));
  }
  __syncthreads();

  int wave = tid >> 5, lane = tid & 31;
  int half = lane >> 4, m = lane & 15, col = lane & 15;
  int colbase = wave << 4;  // 0,16,32,48
  v8f acc = {};
  #pragma unroll
  for (int kt = 0; kt < 4; ++kt) {
    int k0 = kt << 5;
    v16bf a = ldA(&Hs[m][k0 + half * 16], &Hs[m][k0 + half * 16 + 8]);
    v16bf b = ldB(&W2pk[(size_t)((((kt << 2) + wave) << 5) + lane) << 4]);
    acc = __builtin_amdgcn_wmma_f32_16x16x32_bf16(false, a, false, b,
                                                  (short)0, acc, false, false);
  }
  float bb = b2[colbase + col];
  #pragma unroll
  for (int r = 0; r < 8; ++r) {
    int crow = r + half * 8;
    H2[crow][colbase + col] = fmaxf(acc[r] + bb, 0.0f);
  }
  __syncthreads();
  if (tid < 16) {
    float s = b3[0];
    #pragma unroll
    for (int c = 0; c < 64; ++c) s += H2[tid][c] * W3[c];
    out[row0 + tid] = 1.0f / (1.0f + expf(-s));
  }
}

// ---------------------------------------------------------------------------
extern "C" void kernel_launch(void* const* d_in, const int* in_sizes, int n_in,
                              void* d_out, int out_size, void* d_ws, size_t ws_size,
                              hipStream_t stream) {
  (void)in_sizes; (void)n_in; (void)out_size; (void)ws_size;
  const float* coords = (const float*)d_in[0];
  const int*   ei     = (const int*)d_in[1];
  const float* g0_w1  = (const float*)d_in[2];
  const float* g0_b1  = (const float*)d_in[3];
  const float* g0_g1  = (const float*)d_in[4];
  const float* g0_be1 = (const float*)d_in[5];
  const float* g0_w2  = (const float*)d_in[6];
  const float* g0_b2  = (const float*)d_in[7];
  const float* g0_g2  = (const float*)d_in[8];
  const float* g0_be2 = (const float*)d_in[9];
  const float* g0_w3  = (const float*)d_in[10];
  const float* g0_b3  = (const float*)d_in[11];
  const float* eps0   = (const float*)d_in[12];
  const float* bn0_g  = (const float*)d_in[13];
  const float* bn0_b  = (const float*)d_in[14];
  const float* g1_w1  = (const float*)d_in[15];
  const float* g1_b1  = (const float*)d_in[16];
  const float* g1_g1  = (const float*)d_in[17];
  const float* g1_be1 = (const float*)d_in[18];
  const float* g1_w2  = (const float*)d_in[19];
  const float* g1_b2  = (const float*)d_in[20];
  const float* g1_g2  = (const float*)d_in[21];
  const float* g1_be2 = (const float*)d_in[22];
  const float* g1_w3  = (const float*)d_in[23];
  const float* g1_b3  = (const float*)d_in[24];
  const float* eps1   = (const float*)d_in[25];
  const float* bn1_g  = (const float*)d_in[26];
  const float* bn1_b  = (const float*)d_in[27];
  const float* d_w1   = (const float*)d_in[28];
  const float* d_b1   = (const float*)d_in[29];
  const float* d_g    = (const float*)d_in[30];
  const float* d_be   = (const float*)d_in[31];
  const float* d_w2   = (const float*)d_in[32];
  const float* d_b2   = (const float*)d_in[33];
  const float* d_w3   = (const float*)d_in[34];
  const float* d_b3   = (const float*)d_in[35];

  // -------- workspace layout
  float* nodeA = (float*)d_ws;
  float* nodeB = nodeA + (size_t)N_NODES * HID;
  unsigned short* x1bf  = (unsigned short*)(nodeB + (size_t)N_NODES * HID);
  unsigned short* eh1   = x1bf + (size_t)N_NODES * HID;
  unsigned short* pw    = eh1 + (size_t)N_EDGES * HID;   // 32B-aligned by construction
  unsigned short* pw_g0w2 = pw;
  unsigned short* pw_g0w3 = pw + 1 * 16384;
  unsigned short* pw_g1w1 = pw + 2 * 16384;
  unsigned short* pw_g1w2 = pw + 3 * 16384;
  unsigned short* pw_g1w3 = pw + 4 * 16384;
  unsigned short* pw_dw1  = pw + 5 * 16384;              // 256*128
  unsigned short* pw_dw2  = pw_dw1 + 32768;              // 128*64
  float* stats = (float*)(pw_dw2 + 8192);

  const int GN = N_NODES / 16;   // 6250
  const int GE = N_EDGES / 16;   // 100000

  // -------- pack weights to bf16 B-fragment layout (input-only dependency)
  k_wpack<<<64, 256, 0, stream>>>(g0_w2, pw_g0w2, 128, 128);
  k_wpack<<<64, 256, 0, stream>>>(g0_w3, pw_g0w3, 128, 128);
  k_wpack<<<64, 256, 0, stream>>>(g1_w1, pw_g1w1, 128, 128);
  k_wpack<<<64, 256, 0, stream>>>(g1_w2, pw_g1w2, 128, 128);
  k_wpack<<<64, 256, 0, stream>>>(g1_w3, pw_g1w3, 128, 128);
  k_wpack<<<128, 256, 0, stream>>>(d_w1, pw_dw1, 256, 128);  // rows 0..255 only
  k_wpack<<<32, 256, 0, stream>>>(d_w2, pw_dw2, 128, 64);

  // -------- GIN layer 0
  k_zero<<<512, 256, 0, stream>>>(nodeB, (long)N_NODES * 3);
  k_scatter3<<<4096, 256, 0, stream>>>(coords, ei, nodeB);
  k_lin0<<<4096, 256, 0, stream>>>(coords, nodeB, g0_w1, g0_b1, eps0, nodeA);
  k_zero<<<1, 256, 0, stream>>>(stats, 256);
  k_colstats<<<1024, 256, 0, stream>>>(nodeA, N_NODES, stats);
  k_bnrelu<<<4096, 256, 0, stream>>>(nodeA, N_NODES, stats, g0_g1, g0_be1);

  k_gemm128<<<GN, 256, 0, stream>>>(nodeA, pw_g0w2, g0_b2, nodeB);
  k_zero<<<1, 256, 0, stream>>>(stats, 256);
  k_colstats<<<1024, 256, 0, stream>>>(nodeB, N_NODES, stats);
  k_bnrelu<<<4096, 256, 0, stream>>>(nodeB, N_NODES, stats, g0_g2, g0_be2);

  k_gemm128<<<GN, 256, 0, stream>>>(nodeB, pw_g0w3, g0_b3, nodeA);
  k_zero<<<1, 256, 0, stream>>>(stats, 256);
  k_colstats<<<1024, 256, 0, stream>>>(nodeA, N_NODES, stats);
  k_bnrelu<<<4096, 256, 0, stream>>>(nodeA, N_NODES, stats, bn0_g, bn0_b);  // x0

  // -------- GIN layer 1
  k_zero<<<8192, 256, 0, stream>>>(nodeB, (long)N_NODES * HID);
  k_scatter128<<<8192, 256, 0, stream>>>(nodeA, ei, nodeB);
  k_combine<<<4096, 256, 0, stream>>>(nodeA, nodeB, eps1);

  k_gemm128<<<GN, 256, 0, stream>>>(nodeB, pw_g1w1, g1_b1, nodeA);
  k_zero<<<1, 256, 0, stream>>>(stats, 256);
  k_colstats<<<1024, 256, 0, stream>>>(nodeA, N_NODES, stats);
  k_bnrelu<<<4096, 256, 0, stream>>>(nodeA, N_NODES, stats, g1_g1, g1_be1);

  k_gemm128<<<GN, 256, 0, stream>>>(nodeA, pw_g1w2, g1_b2, nodeB);
  k_zero<<<1, 256, 0, stream>>>(stats, 256);
  k_colstats<<<1024, 256, 0, stream>>>(nodeB, N_NODES, stats);
  k_bnrelu<<<4096, 256, 0, stream>>>(nodeB, N_NODES, stats, g1_g2, g1_be2);

  k_gemm128<<<GN, 256, 0, stream>>>(nodeB, pw_g1w3, g1_b3, nodeA);
  k_zero<<<1, 256, 0, stream>>>(stats, 256);
  k_colstats<<<1024, 256, 0, stream>>>(nodeA, N_NODES, stats);
  k_bnrelu<<<4096, 256, 0, stream>>>(nodeA, N_NODES, stats, bn1_g, bn1_b);  // x1

  // x1 -> bf16 for the edge gather
  k_f2bf<<<4096, 256, 0, stream>>>(nodeA, x1bf, (long)N_NODES * HID);

  // -------- edge decoder
  k_dec_gemm1<<<GE, 256, 0, stream>>>(x1bf, coords, ei, pw_dw1,
                                      d_w1 + (size_t)256 * HID, d_b1, eh1);
  k_zero<<<1, 256, 0, stream>>>(stats, 256);
  k_colstats_bf<<<2048, 256, 0, stream>>>(eh1, N_EDGES, stats);
  k_dec_tail<<<GE, 128, 0, stream>>>(eh1, stats, d_g, d_be, pw_dw2, d_b2, d_w3, d_b3,
                                     (float*)d_out);
}
